// GraphEncoder_9663676416578
// MI455X (gfx1250) — matrix operand order repaired and checked by reading, not compiled, not run on hardware
//
#include <hip/hip_runtime.h>

// ---------------------------------------------------------------------------
// GATv2 graph encoder for gfx1250 (MI455X).
// GEMMs use v_wmma_f32_16x16x32_f16. Edge softmax/aggregation via L2 atomics.
// ---------------------------------------------------------------------------

#define N_NODES   50000
#define N_EDGES   400000
#define N_GRAPHS  64
#define HEADS     4
#define GAT_DIM   128
#define HC        (HEADS * GAT_DIM)   // 512
#define NEG_SLOPE 0.2f

typedef __attribute__((ext_vector_type(16))) _Float16 v16h;
typedef __attribute__((ext_vector_type(8)))  _Float16 v8h;
typedef __attribute__((ext_vector_type(8)))  float    v8f;

// order-preserving float<->uint mapping for atomicMax on signed floats
__device__ __forceinline__ unsigned mapf(float f) {
  unsigned u = __float_as_uint(f);
  return (u & 0x80000000u) ? ~u : (u | 0x80000000u);
}
__device__ __forceinline__ float unmapf(unsigned u) {
  return __uint_as_float((u & 0x80000000u) ? (u & 0x7FFFFFFFu) : ~u);
}

// ---------------- layer 0: K=3 naive transform (xl and xr) -----------------
__global__ void gemm_in3_kernel(const float* __restrict__ x,
                                const float* __restrict__ Wl, const float* __restrict__ bl,
                                const float* __restrict__ Wr, const float* __restrict__ br,
                                float* __restrict__ xl, float* __restrict__ xr) {
  long t = (long)blockIdx.x * blockDim.x + threadIdx.x;
  if (t >= (long)N_NODES * HC) return;
  int n = (int)(t >> 9);
  int c = (int)(t & (HC - 1));
  float x0 = x[n * 3 + 0], x1 = x[n * 3 + 1], x2 = x[n * 3 + 2];
  xl[t] = fmaf(x0, Wl[c], fmaf(x1, Wl[HC + c], fmaf(x2, Wl[2 * HC + c], bl[c])));
  xr[t] = fmaf(x0, Wr[c], fmaf(x1, Wr[HC + c], fmaf(x2, Wr[2 * HC + c], br[c])));
}

// ---------------- f32 -> f16 activation convert ----------------------------
__global__ void cvt_f16_kernel(const float* __restrict__ src, _Float16* __restrict__ dst, int n) {
  int t = blockIdx.x * blockDim.x + threadIdx.x;
  if (t < n) dst[t] = (_Float16)src[t];
}

// ---------- pre-swizzle W [128,512] f32 into WMMA B-fragment layout --------
// B frag (32x16 f16, wave32), contiguous K order used here: lanes 0-15 hold K=0..15 (cols=lane),
// lanes 16-31 hold K=16..31 (cols=lane-16)  => k_local = h + 16*(lane>>4)
// packed[(((ntile*4)+ktile)*32+lane)*16 + h] = W[(ktile*32 + 16*(lane>>4) + h)*512 + ntile*16 + (lane&15)]
__global__ void pack_w_kernel(const float* __restrict__ W, _Float16* __restrict__ packed) {
  int t = blockIdx.x * blockDim.x + threadIdx.x;   // 0 .. 65535
  if (t >= 32 * 4 * 32 * 16) return;
  int h    = t & 15;
  int lane = (t >> 4) & 31;
  int kt   = (t >> 9) & 3;
  int nt   = t >> 11;
  int k = kt * 32 + ((lane >> 4) << 4) + h;
  int n = nt * 16 + (lane & 15);
  packed[t] = (_Float16)W[k * HC + n];
}

// ---------------- WMMA GEMM: [M,128] f16 @ [128,512] -> f32 + bias ---------
// block = 256 (8 waves); wave w -> ntile = blockIdx.y*8 + w ; blockIdx.x = mtile
__global__ __launch_bounds__(256)
void gemm_wmma_kernel(const _Float16* __restrict__ A, const _Float16* __restrict__ Bp,
                      const float* __restrict__ bias, float* __restrict__ C) {
  int lane  = threadIdx.x & 31;
  int w     = threadIdx.x >> 5;
  int mtile = blockIdx.x;
  int ntile = blockIdx.y * 8 + w;
  int hs    = lane >> 4;

  const _Float16* arow = A + (size_t)(mtile * 16 + (lane & 15)) * GAT_DIM;
  v8f acc = {};
#pragma unroll
  for (int kt = 0; kt < 4; ++kt) {
    // A fragment: 16x32 f16, contiguous K (lane<16: K 0-15; lane>=16: K 16-31)
    int kb = kt * 32 + hs * 16;
    v8h alo = *(const v8h*)(arow + kb);
    v8h ahi = *(const v8h*)(arow + kb + 8);
    v16h a = __builtin_shufflevector(alo, ahi, 0, 1, 2, 3, 4, 5, 6, 7,
                                               8, 9, 10, 11, 12, 13, 14, 15);
    // B fragment: pre-swizzled, contiguous 32B per lane
    v16h b = *(const v16h*)(Bp + (((size_t)ntile * 4 + kt) * 32 + lane) * 16);
    acc = __builtin_amdgcn_wmma_f32_16x16x32_f16(false, a, false, b,
                                                 (short)0, acc, false, false);
  }
  // C layout: VGPR i -> row mtile*16 + i + 8*hs, col = ntile*16 + (lane&15)
  int col = ntile * 16 + (lane & 15);
  float bv = bias[col];
  float* crow = C + (size_t)(mtile * 16 + hs * 8) * HC + col;
#pragma unroll
  for (int i = 0; i < 8; ++i) crow[(size_t)i * HC] = acc[i] + bv;
}

// ---------------- edge attention scores + segment max ----------------------
// one block (128 thr = 4 waves) per edge; wave = head; lane covers 4 channels
__global__ __launch_bounds__(128)
void edge_score_kernel(const int* __restrict__ ei, const float* __restrict__ xl,
                       const float* __restrict__ xr, const float* __restrict__ att,
                       float* __restrict__ esc, unsigned* __restrict__ emaxU) {
  int e = blockIdx.x;
  int h = threadIdx.x >> 5;
  int lane = threadIdx.x & 31;
  int s = ei[e];
  int d = ei[N_EDGES + e];
  const float* pl = xl + (size_t)s * HC + h * GAT_DIM;
  const float* pr = xr + (size_t)d * HC + h * GAT_DIM;
  const float* pa = att + h * GAT_DIM;
  float acc = 0.f;
#pragma unroll
  for (int j = 0; j < 4; ++j) {
    int c = lane + j * 32;
    float m = pl[c] + pr[c];
    m = (m > 0.f) ? m : NEG_SLOPE * m;
    acc = fmaf(m, pa[c], acc);
  }
#pragma unroll
  for (int off = 16; off; off >>= 1) acc += __shfl_xor(acc, off, 32);
  if (lane == 0) {
    esc[(size_t)e * HEADS + h] = acc;
    atomicMax(&emaxU[(size_t)d * HEADS + h], mapf(acc));
  }
}

// ---------------- exp(e - max) and denominator -----------------------------
__global__ void edge_exp_kernel(const int* __restrict__ ei, float* __restrict__ esc,
                                const unsigned* __restrict__ emaxU, float* __restrict__ denom) {
  int t = blockIdx.x * blockDim.x + threadIdx.x;
  if (t >= N_EDGES * HEADS) return;
  int e = t >> 2, h = t & 3;
  int d = ei[N_EDGES + e];
  float mx = unmapf(emaxU[(size_t)d * HEADS + h]);
  float ee = expf(esc[t] - mx);
  esc[t] = ee;
  atomicAdd(&denom[(size_t)d * HEADS + h], ee);
}

// ---------------- weighted aggregation into agg[dst] -----------------------
// one block (256 thr) per edge; thread t: head = t>>6, channels c and c+64
__global__ __launch_bounds__(256)
void edge_agg_kernel(const int* __restrict__ ei, const float* __restrict__ esc,
                     const float* __restrict__ denom, const float* __restrict__ xl,
                     float* __restrict__ agg) {
  int e = blockIdx.x;
  int h = threadIdx.x >> 6;
  int c = threadIdx.x & 63;
  int s = ei[e];
  int d = ei[N_EDGES + e];
  float alpha = esc[(size_t)e * HEADS + h] / denom[(size_t)d * HEADS + h];
  const float* pl = xl + (size_t)s * HC + h * GAT_DIM;
  float* pd = agg + (size_t)d * HC + h * GAT_DIM;
  atomicAdd(&pd[c], alpha * pl[c]);
  atomicAdd(&pd[c + 64], alpha * pl[c + 64]);
}

// ---------------- head mean + bias + ELU (+ optional graph pooling) --------
__global__ void node_finalize_kernel(const float* __restrict__ agg, const float* __restrict__ bias,
                                     const int* __restrict__ batch, float* __restrict__ hcur,
                                     float* __restrict__ pool, float* __restrict__ cnt, int last) {
  int t = blockIdx.x * blockDim.x + threadIdx.x;
  if (t >= N_NODES * GAT_DIM) return;
  int n = t >> 7, c = t & 127;
  const float* p = agg + (size_t)n * HC;
  float v = 0.25f * (p[c] + p[GAT_DIM + c] + p[2 * GAT_DIM + c] + p[3 * GAT_DIM + c]) + bias[c];
  v = (v > 0.f) ? v : (expf(v) - 1.f);
  hcur[t] = v;
  if (last) {
    int g = batch[n];
    atomicAdd(&pool[g * GAT_DIM + c], v);
    if (c == 0) atomicAdd(&cnt[g], 1.f);
  }
}

// ---------------- final: mean pool divide + linear [128->128] --------------
__global__ void final_linear_kernel(const float* __restrict__ pool, const float* __restrict__ cnt,
                                    const float* __restrict__ W, const float* __restrict__ b,
                                    float* __restrict__ out) {
  int t = blockIdx.x * blockDim.x + threadIdx.x;
  if (t >= N_GRAPHS * GAT_DIM) return;
  int g = t >> 7, j = t & 127;
  float inv = 1.f / fmaxf(cnt[g], 1.f);
  float acc = b[j];
  for (int k = 0; k < GAT_DIM; ++k)
    acc = fmaf(pool[g * GAT_DIM + k] * inv, W[k * GAT_DIM + j], acc);
  out[t] = acc;
}

// ===========================================================================
extern "C" void kernel_launch(void* const* d_in, const int* in_sizes, int n_in,
                              void* d_out, int out_size, void* d_ws, size_t ws_size,
                              hipStream_t stream) {
  (void)in_sizes; (void)n_in; (void)out_size; (void)ws_size;
  const float* x     = (const float*)d_in[0];
  const int*   ei    = (const int*)d_in[1];
  const int*   batch = (const int*)d_in[2];
  const float* Wl[3], *bl[3], *Wr[3], *br[3], *att[3], *bb[3];
  for (int l = 0; l < 3; ++l) {
    Wl[l]  = (const float*)d_in[3 + 6 * l + 0];
    bl[l]  = (const float*)d_in[3 + 6 * l + 1];
    Wr[l]  = (const float*)d_in[3 + 6 * l + 2];
    br[l]  = (const float*)d_in[3 + 6 * l + 3];
    att[l] = (const float*)d_in[3 + 6 * l + 4];
    bb[l]  = (const float*)d_in[3 + 6 * l + 5];
  }
  const float* lin_W = (const float*)d_in[21];
  const float* lin_b = (const float*)d_in[22];

  // ---- workspace layout (floats) ----
  float* f = (float*)d_ws;
  size_t o = 0;
  float*     hcur  = f + o;                 o += (size_t)N_NODES * GAT_DIM;   // 6.4M
  float*     xl    = f + o;                 o += (size_t)N_NODES * HC;        // 25.6M
  float*     xr    = f + o;                 o += (size_t)N_NODES * HC;        // 25.6M (reused as agg)
  float*     esc   = f + o;                 o += (size_t)N_EDGES * HEADS;     // 1.6M
  unsigned*  emaxU = (unsigned*)(f + o);    o += (size_t)N_NODES * HEADS;     // 0.2M
  float*     denom = f + o;                 o += (size_t)N_NODES * HEADS;     // 0.2M
  _Float16*  hf16  = (_Float16*)(f + o);    o += (size_t)N_NODES * GAT_DIM / 2;
  _Float16*  packW = (_Float16*)(f + o);    o += 2 * (size_t)GAT_DIM * HC / 2; // two packed W's
  float*     pool  = f + o;                 o += (size_t)N_GRAPHS * GAT_DIM;
  float*     cnt   = f + o;                 o += N_GRAPHS;

  hipMemsetAsync(pool, 0, (size_t)(N_GRAPHS * GAT_DIM + N_GRAPHS) * sizeof(float), stream);

  const int MTILES = N_NODES / 16;  // 3125 exactly

  for (int l = 0; l < 3; ++l) {
    if (l == 0) {
      long total = (long)N_NODES * HC;
      gemm_in3_kernel<<<(unsigned)((total + 255) / 256), 256, 0, stream>>>(
          x, Wl[0], bl[0], Wr[0], br[0], xl, xr);
    } else {
      pack_w_kernel<<<256, 256, 0, stream>>>(Wl[l], packW);
      pack_w_kernel<<<256, 256, 0, stream>>>(Wr[l], packW + (size_t)GAT_DIM * HC);
      int ncv = N_NODES * GAT_DIM;
      cvt_f16_kernel<<<(ncv + 255) / 256, 256, 0, stream>>>(hcur, hf16, ncv);
      gemm_wmma_kernel<<<dim3(MTILES, 4), 256, 0, stream>>>(hf16, packW, bl[l], xl);
      gemm_wmma_kernel<<<dim3(MTILES, 4), 256, 0, stream>>>(hf16, packW + (size_t)GAT_DIM * HC,
                                                            br[l], xr);
    }
    hipMemsetAsync(emaxU, 0, (size_t)N_NODES * HEADS * sizeof(unsigned), stream);
    hipMemsetAsync(denom, 0, (size_t)N_NODES * HEADS * sizeof(float), stream);

    edge_score_kernel<<<N_EDGES, 128, 0, stream>>>(ei, xl, xr, att[l], esc, emaxU);
    edge_exp_kernel<<<(N_EDGES * HEADS + 255) / 256, 256, 0, stream>>>(ei, esc, emaxU, denom);

    // xr is free now: reuse as aggregation buffer
    hipMemsetAsync(xr, 0, (size_t)N_NODES * HC * sizeof(float), stream);
    edge_agg_kernel<<<N_EDGES, 256, 0, stream>>>(ei, esc, denom, xl, xr);

    node_finalize_kernel<<<(N_NODES * GAT_DIM + 255) / 256, 256, 0, stream>>>(
        xr, bb[l], batch, hcur, pool, cnt, l == 2 ? 1 : 0);
  }

  final_linear_kernel<<<(N_GRAPHS * GAT_DIM + 255) / 256, 256, 0, stream>>>(
      pool, cnt, lin_W, lin_b, (float*)d_out);
}
